// MyCrossAttention_66048007078341
// MI455X (gfx1250) — matrix operand order repaired and checked by reading, compiled, not run on hardware
//
#include <hip/hip_runtime.h>
#include <hip/hip_bf16.h>

typedef __bf16 bf16;
typedef __attribute__((ext_vector_type(16))) __bf16 v16bf;
typedef __attribute__((ext_vector_type(8)))  __bf16 v8bf;
typedef __attribute__((ext_vector_type(8)))  float  v8f;

#define WMMA_BF16(a, b, c) \
  __builtin_amdgcn_wmma_f32_16x16x32_bf16(false, (a), false, (b), (short)0, (c), false, false)

// CDNA5 async global->LDS copy (ASYNCcnt-tracked, bypasses VGPRs).
// lds_addr: byte offset within the workgroup LDS allocation (low 32 bits of a
// generic pointer into __shared__ memory, per the flat-aperture mapping).
__device__ __forceinline__ void async_ld_b128(unsigned lds_addr, const void* gaddr) {
  asm volatile("global_load_async_to_lds_b128 %0, %1, off"
               :: "v"(lds_addr), "v"(gaddr)
               : "memory");
}
__device__ __forceinline__ void wait_async0() {
  asm volatile("s_wait_asynccnt 0x0" ::: "memory");
}

// Build a 16-element bf16 fragment from two runs of 8 f32 values.
__device__ __forceinline__ v16bf frag_from_f32(const float* p0, const float* p1) {
  v16bf r;
#pragma unroll
  for (int j = 0; j < 8; ++j) { r[j] = (__bf16)p0[j]; r[j + 8] = (__bf16)p1[j]; }
  return r;
}

// Build a 16-element bf16 fragment from two runs of 8 bf16 values (16B-aligned).
__device__ __forceinline__ v16bf frag_from_bf16(const bf16* p0, const bf16* p1) {
  v8bf lo = *(const v8bf*)p0;
  v8bf hi = *(const v8bf*)p1;
  v16bf r;
#pragma unroll
  for (int j = 0; j < 8; ++j) { r[j] = lo[j]; r[j + 8] = hi[j]; }
  return r;
}

// ---------------------------------------------------------------------------
// Y[R,256] = X[R,256] @ W[256,256]^T  (f32 in, OutT out), 8 waves per block.
// Each wave computes one 16x16 tile; block tile = 32 rows x 64 cols.
// grid = (R/32, 4), block = 256.
// ---------------------------------------------------------------------------
template <typename OutT>
__global__ void proj_kernel(const float* __restrict__ X, const float* __restrict__ W,
                            OutT* __restrict__ Y) {
  const int lane = threadIdx.x & 31;
  const int wave = threadIdx.x >> 5;
  const int half = lane >> 4;
  const int lc   = lane & 15;
  const int rowTile = blockIdx.x * 32 + (wave >> 2) * 16;
  const int colTile = blockIdx.y * 64 + (wave & 3) * 16;
  const int row = rowTile + lc;   // A-frag row for this lane
  const int col = colTile + lc;   // B-frag column (= W row, since B = W^T)

  v8f c = {};
#pragma unroll
  for (int k0 = 0; k0 < 256; k0 += 32) {
    const float* ap = X + (size_t)row * 256 + k0 + half * 8;
    v16bf a = frag_from_f32(ap, ap + 16);
    const float* bp = W + (size_t)col * 256 + k0 + half * 8;
    v16bf b = frag_from_f32(bp, bp + 16);
    c = WMMA_BF16(a, b, c);
  }
#pragma unroll
  for (int r = 0; r < 8; ++r)
    Y[(size_t)(rowTile + r + 8 * half) * 256 + col] = (OutT)c[r];
}

// ---------------------------------------------------------------------------
// Flash attention with double-buffered LDS tiles and async K staging.
// block = 128 (4 waves), each wave owns a 16-row query tile (block: 64 rows).
// grid = (NQ/64, B). Step = 32 keys. Per step: wait_asynccnt + one barrier,
// fire next tile's async loads, then 32 WMMAs on the current tile.
// ---------------------------------------------------------------------------
__device__ __forceinline__ void stage_tile(int tid, const bf16* __restrict__ K,
                                           const bf16* __restrict__ V,
                                           bf16 (*kSb)[264], bf16 (*vSb)[40],
                                           int kb, int NK) {
  for (int i = tid; i < 1024; i += 128) {
    const int key = i >> 5;        // 0..31
    const int ch  = i & 31;        // 8-element chunk of the 256-dim
    const size_t goff = (size_t)(kb + key) * 256 + ch * 8;
    // K tile: async DMA straight into LDS (row-major; no VGPR data movement).
    async_ld_b128((unsigned)(uintptr_t)&kSb[key][ch * 8], K + goff);
    // V tile: manual path so we can transpose into [dim][key] for PV B-frags.
    v8bf dv = *(const v8bf*)(V + goff);
#pragma unroll
    for (int j = 0; j < 8; ++j) vSb[ch * 8 + j][key] = dv[j];
    if (kb + 64 < NK) {
      __builtin_prefetch((const void*)(K + goff + 64 * 256), 0, 1);
      __builtin_prefetch((const void*)(V + goff + 64 * 256), 0, 1);
    }
  }
}

__global__ void __launch_bounds__(128, 1)
attn_kernel(const bf16* __restrict__ Qh, const bf16* __restrict__ Kh,
            const bf16* __restrict__ Vh, const unsigned char* __restrict__ Msk,
            float* __restrict__ Xh) {
  const int NQ = 2048, NK = 4096;
  __shared__ __align__(16) bf16 qS[64][264];       // 64 q-rows x 256 dims (padded)
  __shared__ __align__(16) bf16 kS[2][32][264];    // double-buffered K tiles
  __shared__ __align__(16) bf16 vS[2][256][40];    // double-buffered V^T tiles
  __shared__ __align__(16) bf16 pS[4][16][40];     // per-wave P re-layout staging

  const int lane = threadIdx.x & 31;
  const int wave = threadIdx.x >> 5;
  const int half = lane >> 4;
  const int lc   = lane & 15;
  const int b  = blockIdx.y;
  const int q0 = blockIdx.x * 64;                  // block's first query row

  const bf16* Q = Qh + ((size_t)b * NQ + q0) * 256;
  const bf16* K = Kh + (size_t)b * NK * 256;
  const bf16* V = Vh + (size_t)b * NK * 256;
  const unsigned char* M = Msk + (size_t)b * NK;

  // Stage the block's 64x256 Q tile into LDS once.
  for (int i = threadIdx.x; i < 2048; i += 128) {  // 64 rows x 32 chunks of 8
    const int row = i >> 5, ch = i & 31;
    *(v8bf*)&qS[row][ch * 8] = *(const v8bf*)(Q + (size_t)row * 256 + ch * 8);
  }

  float m_i[8], l_i[8];
  v8f acc[16];
  v8f vzero = {};
#pragma unroll
  for (int r = 0; r < 8; ++r) { m_i[r] = -3.0e38f; l_i[r] = 0.0f; }
#pragma unroll
  for (int ch = 0; ch < 16; ++ch) acc[ch] = vzero;

  const int qrow = wave * 16 + lc;  // this lane's A-fragment row in qS

  // Prologue: fire async stage of the first tile into buffer 0.
  stage_tile(threadIdx.x, K, V, kS[0], vS[0], 0, NK);

  int buf = 0;
  for (int kb = 0; kb < NK; kb += 32) {
    wait_async0();      // my async K writes for `buf` have landed in LDS
    __syncthreads();    // everyone's staging (async + DS) visible; prior reads done

    // Overlap: launch next tile's staging into the other buffer, then compute.
    if (kb + 32 < NK)
      stage_tile(threadIdx.x, K, V, kS[buf ^ 1], vS[buf ^ 1], kb + 32, NK);

    const bf16 (*kT)[264] = kS[buf];
    const bf16 (*vT)[40]  = vS[buf];

    // S = Q K^T for this 16x32 key slab (two 16x16 C tiles).
    v8f s0 = {}, s1 = {};
#pragma unroll
    for (int ch = 0; ch < 8; ++ch) {
      const bf16* qp = &qS[qrow][ch * 32 + half * 8];
      v16bf qa = frag_from_bf16(qp, qp + 16);
      const bf16* p0 = &kT[lc][ch * 32 + half * 8];
      s0 = WMMA_BF16(qa, frag_from_bf16(p0, p0 + 16), s0);
      const bf16* p1 = &kT[16 + lc][ch * 32 + half * 8];
      s1 = WMMA_BF16(qa, frag_from_bf16(p1, p1 + 16), s1);
    }

    const float SC = 0.0625f;  // 1/sqrt(256)
    const float madd0 = M[kb + lc]      ? 0.0f : -10000.0f;
    const float madd1 = M[kb + 16 + lc] ? 0.0f : -10000.0f;

    // Online softmax per row (row = r + 8*half lives in one 16-lane group).
#pragma unroll
    for (int r = 0; r < 8; ++r) {
      float x0 = s0[r] * SC + madd0;
      float x1 = s1[r] * SC + madd1;
      float mx = fmaxf(x0, x1);
#pragma unroll
      for (int off = 8; off; off >>= 1) mx = fmaxf(mx, __shfl_xor(mx, off, 16));
      const float mnew = fmaxf(m_i[r], mx);
      const float p0 = __expf(x0 - mnew);
      const float p1 = __expf(x1 - mnew);
      float rs = p0 + p1;
#pragma unroll
      for (int off = 8; off; off >>= 1) rs += __shfl_xor(rs, off, 16);
      const float corr = __expf(m_i[r] - mnew);
      l_i[r] = l_i[r] * corr + rs;
      m_i[r] = mnew;
#pragma unroll
      for (int ch = 0; ch < 16; ++ch) acc[ch][r] *= corr;
      // Stash P (C layout) for A-fragment re-layout; per-wave LDS, in-order DS.
      pS[wave][r + 8 * half][lc]      = (__bf16)p0;
      pS[wave][r + 8 * half][16 + lc] = (__bf16)p1;
    }

    // Reload P as a 16x32 A-fragment and accumulate O += P @ V.
    const bf16* pp = &pS[wave][lc][half * 8];
    v16bf pa = frag_from_bf16(pp, pp + 16);
#pragma unroll
    for (int ch = 0; ch < 16; ++ch) {
      const bf16* vp = &vT[ch * 16 + lc][half * 8];
      acc[ch] = WMMA_BF16(pa, frag_from_bf16(vp, vp + 16), acc[ch]);
    }
    buf ^= 1;
  }

  // Normalize and write f32 attention output.
#pragma unroll
  for (int ch = 0; ch < 16; ++ch) {
#pragma unroll
    for (int r = 0; r < 8; ++r) {
      Xh[((size_t)b * NQ + q0 + wave * 16 + r + 8 * half) * 256 + ch * 16 + lc] =
          acc[ch][r] / l_i[r];
    }
  }
}

// ---------------------------------------------------------------------------
extern "C" void kernel_launch(void* const* d_in, const int* in_sizes, int n_in,
                              void* d_out, int out_size, void* d_ws, size_t ws_size,
                              hipStream_t stream) {
  const float* q  = (const float*)d_in[0];
  const float* k  = (const float*)d_in[1];
  const float* v  = (const float*)d_in[2];
  const unsigned char* sm = (const unsigned char*)d_in[3];
  const float* Wq = (const float*)d_in[4];
  const float* Wk = (const float*)d_in[5];
  const float* Wv = (const float*)d_in[6];
  const float* Wp = (const float*)d_in[7];

  // Workspace layout (bytes):
  //   qh  bf16 [8*2048*256] @ 0          (8 MB)
  //   kh  bf16 [8*4096*256] @ 8388608    (16 MB)
  //   vh  bf16 [8*4096*256] @ 25165824   (16 MB)
  //   xh  f32  [8*2048*256] @ 41943040   (16 MB)
  char* ws = (char*)d_ws;
  bf16* qh = (bf16*)(ws);
  bf16* kh = (bf16*)(ws + 8388608);
  bf16* vh = (bf16*)(ws + 25165824);
  float* xh = (float*)(ws + 41943040);

  const dim3 blk(256);
  proj_kernel<__bf16><<<dim3(16384 / 32, 4), blk, 0, stream>>>(q, Wq, qh);
  proj_kernel<__bf16><<<dim3(32768 / 32, 4), blk, 0, stream>>>(k, Wk, kh);
  proj_kernel<__bf16><<<dim3(32768 / 32, 4), blk, 0, stream>>>(v, Wv, vh);

  attn_kernel<<<dim3(2048 / 64, 8), dim3(128), 0, stream>>>(qh, kh, vh, sm, xh);

  proj_kernel<float><<<dim3(16384 / 32, 4), blk, 0, stream>>>(xh, Wp, (float*)d_out);
}
